// Segment_3DCenter_8057358647949
// MI455X (gfx1250) — matrix operand order repaired and checked by reading, not compiled, not run on hardware
//
#include <hip/hip_runtime.h>
#include <hip/hip_bf16.h>

typedef _Float16 f16;
typedef __attribute__((ext_vector_type(16))) _Float16 v16h;
typedef __attribute__((ext_vector_type(8)))  float    v8f;

#define NB   64
#define NC   128
#define NH   11
#define NW   11
#define HW   121
#define NOUT 16

// ---------------------------------------------------------------------------
// x (f32) -> f16 copy, same (B, C=128, 11, 11) layout (C acts as conv3d depth)
// ---------------------------------------------------------------------------
__global__ void k_convert_x(const float* __restrict__ x, f16* __restrict__ xh) {
    int t = blockIdx.x * blockDim.x + threadIdx.x;
    if (t < NB * NC * HW) xh[t] = (f16)x[t];
}

// ---------------------------------------------------------------------------
// softmax of cpr_w rows: (3,16,128) -> soft (3,16,128)
// ---------------------------------------------------------------------------
__global__ void k_softmax(const float* __restrict__ cpr_w, float* __restrict__ soft) {
    int row = blockIdx.x * blockDim.x + threadIdx.x;
    if (row >= 48) return;
    const float* src = cpr_w + row * 128;
    float m = src[0];
    for (int d = 1; d < 128; ++d) m = fmaxf(m, src[d]);
    float sum = 0.f;
    for (int d = 0; d < 128; ++d) sum += expf(src[d] - m);
    float inv = 1.f / sum;
    for (int d = 0; d < 128; ++d) soft[row * 128 + d] = expf(src[d] - m) * inv;
}

// ---------------------------------------------------------------------------
// Pre-pack conv3d weights into WMMA A-fragment lane order (f16).
// A intra-fragment mapping: lane L -> row M=L&15; element e ->
//   K = (L>>4)*16 + chunk*32 + e.
// K semantics:
//   conv1 (K=63, scatter path): K = tap (kd*9+kh*3+kw), single channel.
//   conv2/3 (FAST path):        K = chunk*32 + tap_sel*16 + channel,
//                               tap = chunk*2 + tap_sel  (63 taps + 1 pad).
// Layout: wp[((chunk*32+lane)*16)+e].
// Regions: [0,1024): conv1; [1024,17408): conv2; [17408,33792): conv3.
// ---------------------------------------------------------------------------
__global__ void k_pack(const float* __restrict__ w3d1,
                       const float* __restrict__ w3d23,
                       f16* __restrict__ wp) {
    int t = blockIdx.x * blockDim.x + threadIdx.x;
    if (t >= 33792) return;
    const float* src; int rem; bool fast;
    if (t < 1024)       { src = w3d1;          fast = false; rem = t;         }
    else if (t < 17408) { src = w3d23;         fast = true;  rem = t - 1024;  }
    else                { src = w3d23 + 16128; fast = true;  rem = t - 17408; }
    int e     = rem & 15;
    int lane  = (rem >> 4) & 31;
    int chunk = rem >> 9;
    int M = lane & 15;
    int K = (lane >> 4) * 16 + chunk * 32 + e;
    float v = 0.f;
    if (fast) {
        int ch  = K & 15;
        int ts  = (K >> 4) & 1;
        int ck  = K >> 5;
        int tap = ck * 2 + ts;
        if (tap < 63) v = src[(M * 16 + ch) * 63 + tap];
    } else {
        if (K < 63) v = src[M * 63 + K];
    }
    wp[t] = (f16)v;
}

// ---------------------------------------------------------------------------
// WMMA conv3d:  out[b,o,d,h,w] = relu(bias[o] + sum_{i,kd,kh,kw}
//                 in[b,i,d+kd-3,h+kh-1,w+kw-1] * W[o,i,kd,kh,kw])
// in/out f16 (B, CH, 128, 121).  One workgroup = (batch, d-block).
// FAST (CH==16): halo staged channel-major so each lane's 16 B-fragment
// elements (16 consecutive K = 16 channels of one tap) are one contiguous
// 32B LDS vector load -> 2x ds_load_b128 per v_wmma.
// !FAST (CH==1): u16 K->offset LUT scatter path.
// ---------------------------------------------------------------------------
template <int CH, int DBLK, int KTOT, int NCHUNK, bool FAST>
__global__ __launch_bounds__(128) void k_conv3d(
    const f16*  __restrict__ in,
    const f16*  __restrict__ wpack,
    const float* __restrict__ bias,
    f16*        __restrict__ out) {
    constexpr int DS    = DBLK + 6;
    constexpr int SP    = DS * 169;          // spatial halo elements
    constexpr int NPOS  = DBLK * HW;
    constexpr int NTILE = (NPOS + 15) / 16;
    constexpr int NDB   = 128 / DBLK;
    constexpr int KTOTP = NCHUNK * 32;
    __shared__ alignas(32) f16 s_in[CH * SP];
    __shared__ unsigned short s_koff[KTOTP];
    __shared__ unsigned short s_toff[64];

    const int b   = blockIdx.x / NDB;
    const int d0  = (blockIdx.x % NDB) * DBLK;
    const int tid = threadIdx.x;
    constexpr int CSTR = 128 * HW;           // per-channel stride in `in`

    if (FAST) {
        // tap -> spatial LDS offset LUT (tap 63 = padding -> 0; A is 0 there)
        for (int t = tid; t < 64; t += 128) {
            int off = 0;
            if (t < 63) {
                int kd = t / 9, r = t % 9, kh = r / 3, kw = r % 3;
                off = (kd * 13 + kh) * 13 + kw;
            }
            s_toff[t] = (unsigned short)off;
        }
        // channel-major halo: s_in[sp*CH + ch]
        for (int sp = tid; sp < SP; sp += 128) {
            int wz = sp % 13; int tt = sp / 13;
            int hz = tt % 13; int dz = tt / 13;
            int d = d0 - 3 + dz, h = hz - 1, w = wz - 1;
            f16 tmp[CH];
            if (d >= 0 && d < 128 && (unsigned)h < 11u && (unsigned)w < 11u) {
                int gbase = (b * CH * 128 + d) * HW + h * 11 + w;
#pragma unroll
                for (int ch = 0; ch < CH; ++ch) tmp[ch] = in[gbase + ch * CSTR];
            } else {
#pragma unroll
                for (int ch = 0; ch < CH; ++ch) tmp[ch] = (f16)0.f;
            }
#pragma unroll
            for (int ch = 0; ch < CH; ++ch) s_in[sp * CH + ch] = tmp[ch];
        }
    } else {
        // K -> LDS offset LUT (u16); K >= KTOT maps to 0 (A fragment is 0).
        for (int K = tid; K < KTOTP; K += 128) {
            unsigned short off = 0;
            if (K < KTOT) {
                int kd = K / 9;
                int r2 = K - kd * 9;
                int kh = r2 / 3;
                int kw = r2 - kh * 3;
                off = (unsigned short)((kd * 13 + kh) * 13 + kw);
            }
            s_koff[K] = off;
        }
        for (int idx = tid; idx < SP; idx += 128) {
            int wz = idx % 13; int tt = idx / 13;
            int hz = tt % 13;  int dz = tt / 13;
            int d = d0 - 3 + dz, h = hz - 1, w = wz - 1;
            f16 v = (f16)0.f;
            if (d >= 0 && d < 128 && (unsigned)h < 11u && (unsigned)w < 11u)
                v = in[(b * CH * 128 + d) * HW + h * 11 + w];
            s_in[idx] = v;
        }
    }
    __syncthreads();

    const int lane = tid & 31;
    const int wave = tid >> 5;
    const int col  = lane & 15;          // B-fragment column (N)
    const int ts   = lane >> 4;          // B-fragment K-half selector
    const int kb2  = ts * 16;

    for (int tile = wave; tile < NTILE; tile += 4) {
        int p  = tile * 16 + col;
        int pc = p < NPOS ? p : NPOS - 1;           // clamp for addressing only
        int dloc = pc / HW, hw = pc % HW;
        int h = hw / 11, w = hw % 11;
        int posbase = (dloc * 13 + h) * 13 + w;     // spatial LDS base

        v8f acc = {};
        for (int chunk = 0; chunk < NCHUNK; ++chunk) {
            v16h af = *(const v16h*)(wpack + (chunk * 32 + lane) * 16);
            if (chunk + 1 < NCHUNK)  // gfx1250 global_prefetch_b8
                __builtin_prefetch((const void*)(wpack + ((chunk + 1) * 32 + lane) * 16), 0, 3);
            v16h bf;
            if (FAST) {
                int tap = 2 * chunk + ts;
                bf = *(const v16h*)(s_in + ((int)s_toff[tap] + posbase) * CH);
            } else {
                const int kstart = kb2 + chunk * 32;
#pragma unroll
                for (int e = 0; e < 16; ++e)
                    bf[e] = s_in[(int)s_koff[kstart + e] + posbase];
            }
            acc = __builtin_amdgcn_wmma_f32_16x16x32_f16(
                false, af, false, bf, (short)0, acc, false, false);
        }

        // D layout: element v -> row M = v + 8*(lane>>4), col N = lane&15
        int rowbase = ts * 8;
#pragma unroll
        for (int v = 0; v < 8; ++v) {
            int M = rowbase + v;
            if (p < NPOS) {
                int d = d0 + dloc;
                float y = acc[v] + bias[M];
                out[((b * 16 + M) * 128 + d) * HW + hw] = (f16)(y > 0.f ? y : 0.f);
            }
        }
    }
}

// ---------------------------------------------------------------------------
// cpr: out[b,o,hw] = relu( sum_d soft[o,d] * sa[b,o,d,hw] )
// ---------------------------------------------------------------------------
__global__ void k_cpr(const f16* __restrict__ sa, const float* __restrict__ soft,
                      float* __restrict__ out) {
    int tid = blockIdx.x * blockDim.x + threadIdx.x;
    if (tid >= NB * 16 * HW) return;
    int hw = tid % HW; int t = tid / HW;
    int o = t % 16; int b = t / 16;
    const f16*  p  = sa + (size_t)((b * 16 + o) * 128) * HW + hw;
    const float* sw = soft + o * 128;
    float acc = 0.f;
    for (int d = 0; d < 128; ++d) acc += sw[d] * (float)p[d * HW];
    out[tid] = acc > 0.f ? acc : 0.f;
}

// ---------------------------------------------------------------------------
// 16->16 3x3 pad-1 conv2d + relu over (b,16,11,11); one block per batch.
// ---------------------------------------------------------------------------
__global__ __launch_bounds__(128) void k_conv2d16(
    const float* __restrict__ in, const float* __restrict__ wt,
    const float* __restrict__ bias, float* __restrict__ out) {
    __shared__ float s_f[16 * HW];
    int b = blockIdx.x;
    for (int i = threadIdx.x; i < 16 * HW; i += 128) s_f[i] = in[b * 16 * HW + i];
    __syncthreads();
    for (int task = threadIdx.x; task < 16 * HW; task += 128) {
        int hw = task % HW, o = task / HW;
        int h = hw / 11, w = hw % 11;
        float acc = bias[o];
        for (int i = 0; i < 16; ++i)
            for (int kh = 0; kh < 3; ++kh) {
                int hh = h + kh - 1; if ((unsigned)hh >= 11u) continue;
                for (int kw = 0; kw < 3; ++kw) {
                    int ww = w + kw - 1; if ((unsigned)ww >= 11u) continue;
                    acc += wt[(o * 16 + i) * 9 + kh * 3 + kw] * s_f[i * HW + hh * 11 + ww];
                }
            }
        out[b * 16 * HW + task] = acc > 0.f ? acc : 0.f;
    }
}

// ---------------------------------------------------------------------------
// Per (feat, batch): seg conv -> top-5 -> similar_sum -> center-only ratio
// convs (dil 1 then dil 2, evaluated analytically at pixel (64,64)) -> both
// sideouts.  blockIdx.x = feat*64 + batch.
// ---------------------------------------------------------------------------
__global__ __launch_bounds__(128) void k_sides(
    const float* __restrict__ feats,   // (5, 64, 16, 121)
    const float* __restrict__ x,       // (64, 128, 121)
    const float* __restrict__ seg_w, const float* __restrict__ seg_b,
    const float* __restrict__ so_w,  const float* __restrict__ so_b,
    const float* __restrict__ sp_wa, const float* __restrict__ sp_ba,
    const float* __restrict__ sp_wb, const float* __restrict__ sp_bb,
    const float* __restrict__ sp_sw, const float* __restrict__ sp_sb,
    float* __restrict__ sa_side, float* __restrict__ sp_side) {
    __shared__ float s_f[16 * HW];
    __shared__ float s_act[HW];
    __shared__ float s_diff[HW];
    __shared__ int   s_idx[5];
    __shared__ float s_s[128];
    __shared__ float s_xc[128];
    __shared__ float s_y1[9][16];
    __shared__ float s_y2[16];

    int i = blockIdx.x >> 6;
    int b = blockIdx.x & 63;
    int tid = threadIdx.x;

    const float* f = feats + (size_t)(i * 64 + b) * 16 * HW;
    for (int t = tid; t < 16 * HW; t += 128) s_f[t] = f[t];
    __syncthreads();

    // act = conv2d(feat, seg_w[i]) + seg_b[i]   (1 output channel)
    if (tid < HW) {
        int h = tid / 11, w = tid % 11;
        float acc = seg_b[i];
        for (int ch = 0; ch < 16; ++ch)
            for (int kh = 0; kh < 3; ++kh) {
                int hh = h + kh - 1; if ((unsigned)hh >= 11u) continue;
                for (int kw = 0; kw < 3; ++kw) {
                    int ww = w + kw - 1; if ((unsigned)ww >= 11u) continue;
                    acc += seg_w[(i * 16 + ch) * 9 + kh * 3 + kw] * s_f[ch * HW + hh * 11 + ww];
                }
            }
        s_act[tid] = acc;
    }
    __syncthreads();
    if (tid < HW) s_diff[tid] = fabsf(s_act[tid] - s_act[60]);
    __syncthreads();
    if (tid == 0) {  // top-5 smallest |diff|, lowest index wins ties
        for (int j = 0; j < 5; ++j) {
            float best = 3.0e38f; int bi = 0;
            for (int t = 0; t < HW; ++t)
                if (s_diff[t] < best) { best = s_diff[t]; bi = t; }
            s_idx[j] = bi; s_diff[bi] = 3.0e38f;
        }
    }
    __syncthreads();
    // similar_sum and center slice of x
    {
        const float* xp = x + (size_t)(b * 128 + tid) * HW;
        float s = 0.f;
        for (int j = 0; j < 5; ++j) s += xp[s_idx[j]];
        s_s[tid]  = s;
        s_xc[tid] = xp[60];
    }
    __syncthreads();
    // y1 (dil=1 conv of ratio image) at the 9 positions {62,64,66}^2
    for (int task = tid; task < 144; task += 128) {
        int pos = task >> 4, ch = task & 15;
        int pr = 62 + 2 * (pos / 3), pc = 62 + 2 * (pos % 3);
        float acc = sp_ba[i * 16 + ch];
        for (int kh = 0; kh < 3; ++kh)
            for (int kw = 0; kw < 3; ++kw) {
                int r = pr + kh - 1, c = pc + kw - 1;
                float d0 = s_s[c];  if (fabsf(d0) < 0.01f) d0 = 0.01f;
                float d1 = s_xc[c]; if (fabsf(d1) < 0.01f) d1 = 0.01f;
                float img0 = s_s[r] / d0;
                float img1 = s_xc[r] / d1;
                const float* wa = sp_wa + ((i * 16 + ch) * 2) * 9 + kh * 3 + kw;
                acc += wa[0] * img0 + wa[9] * img1;
            }
        s_y1[pos][ch] = acc > 0.f ? acc : 0.f;
    }
    __syncthreads();
    // y2 = relu(dil=2 conv) at center (64,64)
    if (tid < 16) {
        float acc = sp_bb[i * 16 + tid];
        for (int ii = 0; ii < 16; ++ii)
            for (int kh = 0; kh < 3; ++kh)
                for (int kw = 0; kw < 3; ++kw)
                    acc += sp_wb[((i * 16 + tid) * 16 + ii) * 9 + kh * 3 + kw] *
                           s_y1[kh * 3 + kw][ii];
        s_y2[tid] = acc > 0.f ? acc : 0.f;
    }
    __syncthreads();
    if (tid < 16) {
        float accp = sp_sb[i * 16 + tid];
        float acca = so_b[i * 16 + tid];
        for (int o = 0; o < 16; ++o) {
            accp += sp_sw[(i * 16 + tid) * 16 + o] * s_y2[o];
            acca += so_w[(i * 16 + tid) * 16 + o] * s_f[o * HW + 60];
        }
        sp_side[(i * 64 + b) * 16 + tid] = accp;
        sa_side[(i * 64 + b) * 16 + tid] = acca;
    }
}

// ---------------------------------------------------------------------------
// total = sum_i fuse[i]*sa_side_i + sum_i fuse[5+i]*sp_side_i  (ref order)
// ---------------------------------------------------------------------------
__global__ void k_fuse(const float* __restrict__ sa_side,
                       const float* __restrict__ sp_side,
                       const float* __restrict__ fuse, float* __restrict__ out) {
    int t = blockIdx.x * blockDim.x + threadIdx.x;
    if (t >= NB * 16) return;
    int b = t / 16, j = t % 16;
    float acc = 0.f;
    for (int i = 0; i < 5; ++i) acc += fuse[i] * sa_side[(i * 64 + b) * 16 + j];
    for (int i = 0; i < 5; ++i) acc += fuse[5 + i] * sp_side[(i * 64 + b) * 16 + j];
    out[t] = acc;
}

// ---------------------------------------------------------------------------
static inline size_t al256(size_t x) { return (x + 255) & ~(size_t)255; }

extern "C" void kernel_launch(void* const* d_in, const int* in_sizes, int n_in,
                              void* d_out, int out_size, void* d_ws, size_t ws_size,
                              hipStream_t stream) {
    const float* x      = (const float*)d_in[0];
    const float* w3d1   = (const float*)d_in[1];
    const float* b3d1   = (const float*)d_in[2];
    const float* w3d23  = (const float*)d_in[3];
    const float* b3d23  = (const float*)d_in[4];
    const float* cpr_w  = (const float*)d_in[5];
    const float* w2d45  = (const float*)d_in[6];
    const float* b2d45  = (const float*)d_in[7];
    const float* seg_w  = (const float*)d_in[8];
    const float* seg_b  = (const float*)d_in[9];
    const float* so_w   = (const float*)d_in[10];
    const float* so_b   = (const float*)d_in[11];
    const float* sp_wa  = (const float*)d_in[12];
    const float* sp_ba  = (const float*)d_in[13];
    const float* sp_wb  = (const float*)d_in[14];
    const float* sp_bb  = (const float*)d_in[15];
    const float* sp_sw  = (const float*)d_in[16];
    const float* sp_sb  = (const float*)d_in[17];
    const float* fuse   = (const float*)d_in[18];
    float* out = (float*)d_out;

    // Workspace carve-up (all 256B aligned)
    char* base = (char*)d_ws;
    size_t off = 0;
    f16* xh   = (f16*)(base + off); off += al256((size_t)NB * NC * HW * 2);
    f16* bufA = (f16*)(base + off); off += al256((size_t)NB * 16 * 128 * HW * 2);
    f16* bufB = (f16*)(base + off); off += al256((size_t)NB * 16 * 128 * HW * 2);
    f16* wp   = (f16*)(base + off); off += al256((size_t)33792 * 2);
    float* soft    = (float*)(base + off); off += al256((size_t)3 * 16 * 128 * 4);
    float* feats   = (float*)(base + off); off += al256((size_t)5 * NB * 16 * HW * 4);
    float* sa_side = (float*)(base + off); off += al256((size_t)5 * NB * 16 * 4);
    float* sp_side = (float*)(base + off); off += al256((size_t)5 * NB * 16 * 4);
    (void)off; (void)ws_size; (void)in_sizes; (void)n_in; (void)out_size;

    const size_t slot = (size_t)NB * 16 * HW;  // per-feat floats

    k_convert_x<<<(NB * NC * HW + 255) / 256, 256, 0, stream>>>(x, xh);
    k_softmax<<<1, 64, 0, stream>>>(cpr_w, soft);
    k_pack<<<(33792 + 255) / 256, 256, 0, stream>>>(w3d1, w3d23, wp);

    // conv3d #1: 1 input ch, K=63 -> 2 WMMA chunks (scatter path)
    k_conv3d<1, 16, 63, 2, false><<<NB * (128 / 16), 128, 0, stream>>>(xh, wp, b3d1, bufA);
    k_cpr<<<(int)((slot + 255) / 256), 256, 0, stream>>>(bufA, soft, feats);

    // conv3d #2: 16 ch, 63 taps -> 32 WMMA chunks (vector-B fast path)
    k_conv3d<16, 4, 1008, 32, true><<<NB * 32, 128, 0, stream>>>(bufA, wp + 1024, b3d23, bufB);
    k_cpr<<<(int)((slot + 255) / 256), 256, 0, stream>>>(bufB, soft + 16 * 128, feats + slot);

    // conv3d #3 (ping-pong back into bufA; sa1 already consumed)
    k_conv3d<16, 4, 1008, 32, true><<<NB * 32, 128, 0, stream>>>(bufB, wp + 17408, b3d23 + 16, bufA);
    k_cpr<<<(int)((slot + 255) / 256), 256, 0, stream>>>(bufA, soft + 2 * 16 * 128, feats + 2 * slot);

    // sa4_2d, sa5_2d
    k_conv2d16<<<NB, 128, 0, stream>>>(feats + 2 * slot, w2d45, b2d45, feats + 3 * slot);
    k_conv2d16<<<NB, 128, 0, stream>>>(feats + 3 * slot, w2d45 + 16 * 16 * 9, b2d45 + 16,
                                       feats + 4 * slot);

    k_sides<<<5 * NB, 128, 0, stream>>>(feats, x, seg_w, seg_b, so_w, so_b,
                                        sp_wa, sp_ba, sp_wb, sp_bb, sp_sw, sp_sb,
                                        sa_side, sp_side);
    k_fuse<<<(NB * 16 + 255) / 256, 256, 0, stream>>>(sa_side, sp_side, fuse, out);
}
